// DirectContractedVoxGO_Sto_74732430950534
// MI455X (gfx1250) — matrix operand order repaired and checked by reading, not compiled, not run on hardware
//
#include <hip/hip_runtime.h>
#include <hip/hip_bf16.h>
#include <math.h>

typedef __attribute__((ext_vector_type(16))) _Float16 v16h;
typedef __attribute__((ext_vector_type(8)))  _Float16 v8h;
typedef __attribute__((ext_vector_type(8)))  float    v8f;

#define N_RAYS    4096
#define N_SAMPLES 256
#define M_TOTAL   (N_RAYS * N_SAMPLES)
#define KEPS      12
#define GRIDN     160
#define G3        (GRIDN * GRIDN * GRIDN)
#define K0_DIM    12
#define PE        4
#define WIDTH     128
#define DIM0      39
#define DIM0P     64          // padded K for layer 0
#define NOUTP     16          // padded N for layer 2 (6 -> 16)
#define MTILE     256         // samples per block in fused kernel
#define BG_LEN    0.2f
#define XYZ_MIN   (-1.2f)
#define XYZ_MAX   (1.2f)
#define ACT_SHIFT (-9.2102403f)   // log(1/(1-1e-4) - 1)
#define VOXEL     ((XYZ_MAX - XYZ_MIN) / (float)GRIDN)
#define STEPDIST  (0.5f * VOXEL)
#define NEARV     0.1f
#define INTERVAL  0.5f

__device__ __forceinline__ float softplusf(float x) {
    return (x > 20.f) ? x : log1pf(expf(x));
}
__device__ __forceinline__ float sigmoidf(float x) {
    return 1.f / (1.f + expf(-x));
}

// ---------------------------------------------------------------------------
// Kernel 0: transpose + pad weights to f16 [N][K] layout for WMMA B fragments
// ---------------------------------------------------------------------------
__global__ void prep_weights(const float* __restrict__ w0, const float* __restrict__ w1,
                             const float* __restrict__ w2,
                             _Float16* __restrict__ w0t, _Float16* __restrict__ w1t,
                             _Float16* __restrict__ w2t) {
    int i = blockIdx.x * blockDim.x + threadIdx.x;
    if (i < WIDTH * DIM0P) {                        // w0t: [128][64]
        int n = i / DIM0P, k = i % DIM0P;
        w0t[i] = (k < DIM0) ? (_Float16)w0[k * WIDTH + n] : (_Float16)0.f;
    } else if (i < WIDTH * DIM0P + WIDTH * WIDTH) { // w1t: [128][128]
        int j = i - WIDTH * DIM0P;
        int n = j / WIDTH, k = j % WIDTH;
        w1t[j] = (_Float16)w1[k * WIDTH + n];
    } else if (i < WIDTH * DIM0P + WIDTH * WIDTH + NOUTP * WIDTH) { // w2t: [16][128]
        int j = i - WIDTH * DIM0P - WIDTH * WIDTH;
        int n = j / WIDTH, k = j % WIDTH;
        w2t[j] = (n < 6) ? (_Float16)w2[k * 6 + n] : (_Float16)0.f;
    }
}

// ---------------------------------------------------------------------------
// WMMA fragment loaders.
// A (16x32 f16): lanes 0-15 rows, K 0-7 & 16-23; lanes 16-31 same rows, K 8-15 & 24-31.
// B (32x16 f16): lanes 0-15 = cols, K 0-7 & 16-23; lanes 16-31 same cols, K 8-15 & 24-31.
// ---------------------------------------------------------------------------
__device__ __forceinline__ v16h frag_a(const _Float16* row, int ks, int lhalf) {
    const _Float16* p = row + ks * 32 + lhalf * 8;
    v8h lo = *(const v8h*)p;
    v8h hi = *(const v8h*)(p + 16);
    return __builtin_shufflevector(lo, hi, 0,1,2,3,4,5,6,7,8,9,10,11,12,13,14,15);
}
__device__ __forceinline__ v16h frag_b(const _Float16* colrow, int ks, int lhalf) {
    const _Float16* p = colrow + ks * 32 + lhalf * 8;
    v8h lo = *(const v8h*)p;
    v8h hi = *(const v8h*)(p + 16);
    return __builtin_shufflevector(lo, hi, 0,1,2,3,4,5,6,7,8,9,10,11,12,13,14,15);
}

// ---------------------------------------------------------------------------
// Fused kernel: per 256-sample tile
//   Phase 1 (all 256 threads): contraction + trilinear gathers + PE embedding,
//            feature rows written to LDS (f16), density/std to global.
//   Phase 2 (8 waves x 2 stripes): 3-layer MLP entirely on WMMA.
//            A fragments come from LDS (ds_load_b128); B from L2-resident
//            transposed f16 weights; activations staged in a per-wave LDS slice.
// No feature HBM round-trip.
// ---------------------------------------------------------------------------
__launch_bounds__(256)
__global__ void fused_mlp_kernel(const float* __restrict__ rays_o, const float* __restrict__ rays_d,
                                 const float* __restrict__ dgrid, const float* __restrict__ sgrid,
                                 const float* __restrict__ k0grid,
                                 const _Float16* __restrict__ w0t, const _Float16* __restrict__ w1t,
                                 const _Float16* __restrict__ w2t,
                                 const float* __restrict__ b0, const float* __restrict__ b1,
                                 const float* __restrict__ b2,
                                 float* __restrict__ den, float* __restrict__ stdv,
                                 float* __restrict__ out6) {
    __shared__ _Float16 ftile[MTILE * DIM0P];      // 32 KB feature tile
    __shared__ _Float16 hbuf[8][16 * WIDTH];       // 32 KB per-wave activation tiles

    const int tid   = threadIdx.x;
    const int mbase = blockIdx.x * MTILE;

    // ---------------- Phase 1: features ----------------
    {
        int idx = mbase + tid;
        int r = idx >> 8;
        int s = idx & 255;

        float ox = rays_o[r * 3 + 0], oy = rays_o[r * 3 + 1], oz = rays_o[r * 3 + 2];
        float dx = rays_d[r * 3 + 0], dy = rays_d[r * 3 + 1], dz = rays_d[r * 3 + 2];
        float inv = rsqrtf(dx * dx + dy * dy + dz * dz);
        dx *= inv; dy *= inv; dz *= inv;

        float t = NEARV + STEPDIST * (float)s;
        float px = ox + dx * t, py = oy + dy * t, pz = oz + dz * t;

        float nrm = fmaxf(fabsf(px), fmaxf(fabsf(py), fabsf(pz)));
        if (nrm > 1.0f) {
            float sc = (1.0f + BG_LEN - BG_LEN / nrm) / nrm;
            px *= sc; py *= sc; pz *= sc;
        }

        const float u2g = (float)(GRIDN - 1) / (XYZ_MAX - XYZ_MIN);
        float ux = (px - XYZ_MIN) * u2g;
        float uy = (py - XYZ_MIN) * u2g;
        float uz = (pz - XYZ_MIN) * u2g;
        int x0 = (int)fminf(fmaxf(floorf(ux), 0.f), (float)(GRIDN - 2));
        int y0 = (int)fminf(fmaxf(floorf(uy), 0.f), (float)(GRIDN - 2));
        int z0 = (int)fminf(fmaxf(floorf(uz), 0.f), (float)(GRIDN - 2));
        float fx = ux - (float)x0, fy = uy - (float)y0, fz = uz - (float)z0;
        float gx = 1.f - fx, gy = 1.f - fy, gz = 1.f - fz;

        int o000 = (x0 * GRIDN + y0) * GRIDN + z0;
        int o100 = o000 + GRIDN * GRIDN;
        int o010 = o000 + GRIDN;
        int o001 = o000 + 1;
        int o110 = o100 + GRIDN;
        int o101 = o100 + 1;
        int o011 = o010 + 1;
        int o111 = o110 + 1;

        float w000 = gx * gy * gz, w100 = fx * gy * gz, w010 = gx * fy * gz, w001 = gx * gy * fz;
        float w110 = fx * fy * gz, w101 = fx * gy * fz, w011 = gx * fy * fz, w111 = fx * fy * fz;

        float raw_den = w000 * dgrid[o000] + w100 * dgrid[o100] + w010 * dgrid[o010] +
                        w001 * dgrid[o001] + w110 * dgrid[o110] + w101 * dgrid[o101] +
                        w011 * dgrid[o011] + w111 * dgrid[o111];
        float raw_std = w000 * sgrid[o000] + w100 * sgrid[o100] + w010 * sgrid[o010] +
                        w001 * sgrid[o001] + w110 * sgrid[o110] + w101 * sgrid[o101] +
                        w011 * sgrid[o011] + w111 * sgrid[o111];
        den[idx]  = raw_den;
        stdv[idx] = softplusf(raw_std);

        float fv[DIM0P];
        #pragma unroll
        for (int c = 0; c < K0_DIM; ++c) {
            const float* g = k0grid + (size_t)c * G3;
            fv[c] = w000 * g[o000] + w100 * g[o100] + w010 * g[o010] + w001 * g[o001] +
                    w110 * g[o110] + w101 * g[o101] + w011 * g[o011] + w111 * g[o111];
        }
        fv[12] = dx; fv[13] = dy; fv[14] = dz;
        float vdv[3] = {dx, dy, dz};
        #pragma unroll
        for (int d = 0; d < 3; ++d) {
            #pragma unroll
            for (int f = 0; f < PE; ++f) {
                float ang = vdv[d] * (float)(1 << f);
                fv[15 + d * PE + f] = sinf(ang);
                fv[27 + d * PE + f] = cosf(ang);
            }
        }
        #pragma unroll
        for (int j = DIM0; j < DIM0P; ++j) fv[j] = 0.f;

        _Float16* fr = ftile + tid * DIM0P;
        #pragma unroll
        for (int j = 0; j < DIM0P; ++j) fr[j] = (_Float16)fv[j];
    }

    __syncthreads();

    // ---------------- Phase 2: WMMA MLP ----------------
    const int wave  = tid >> 5;
    const int lane  = tid & 31;
    const int lhalf = lane >> 4;      // 0/1: K sub-range of fragment
    const int lidx  = lane & 15;      // row (A) / col (B,D)
    _Float16* h = hbuf[wave];

    #pragma unroll
    for (int sub = 0; sub < 2; ++sub) {
        const int rowbase = wave * 32 + sub * 16;   // stripe base within tile

        v8f acc[8];
        #pragma unroll
        for (int n = 0; n < 8; ++n)
            #pragma unroll
            for (int e = 0; e < 8; ++e) acc[n][e] = 0.f;

        // ---- layer 0: ftile[16x64] @ W0[64x128] ----
        #pragma unroll
        for (int ks = 0; ks < 2; ++ks) {
            v16h a = frag_a(ftile + (rowbase + lidx) * DIM0P, ks, lhalf);
            #pragma unroll
            for (int n = 0; n < 8; ++n) {
                v16h b = frag_b(w0t + (n * 16 + lidx) * DIM0P, ks, lhalf);
                acc[n] = __builtin_amdgcn_wmma_f32_16x16x32_f16(false, a, false, b,
                                                                (short)0, acc[n], false, false);
            }
        }
        #pragma unroll
        for (int n = 0; n < 8; ++n) {
            int col = n * 16 + lidx;
            float bias = b0[col];
            #pragma unroll
            for (int rr = 0; rr < 8; ++rr) {
                float v = acc[n][rr] + bias;
                h[(rr + 8 * lhalf) * WIDTH + col] = (_Float16)fmaxf(v, 0.f);
            }
        }

        // ---- layer 1: H1[16x128] @ W1[128x128] ----
        #pragma unroll
        for (int n = 0; n < 8; ++n)
            #pragma unroll
            for (int e = 0; e < 8; ++e) acc[n][e] = 0.f;
        #pragma unroll
        for (int ks = 0; ks < 4; ++ks) {
            v16h a = frag_a(h + lidx * WIDTH, ks, lhalf);
            #pragma unroll
            for (int n = 0; n < 8; ++n) {
                v16h b = frag_b(w1t + (n * 16 + lidx) * WIDTH, ks, lhalf);
                acc[n] = __builtin_amdgcn_wmma_f32_16x16x32_f16(false, a, false, b,
                                                                (short)0, acc[n], false, false);
            }
        }
        // same-wave LDS is in-order: all reads above precede these writes
        #pragma unroll
        for (int n = 0; n < 8; ++n) {
            int col = n * 16 + lidx;
            float bias = b1[col];
            #pragma unroll
            for (int rr = 0; rr < 8; ++rr) {
                float v = acc[n][rr] + bias;
                h[(rr + 8 * lhalf) * WIDTH + col] = (_Float16)fmaxf(v, 0.f);
            }
        }

        // ---- layer 2: H2[16x128] @ W2[128x16] ----
        v8f acc2;
        #pragma unroll
        for (int e = 0; e < 8; ++e) acc2[e] = 0.f;
        #pragma unroll
        for (int ks = 0; ks < 4; ++ks) {
            v16h a = frag_a(h + lidx * WIDTH, ks, lhalf);
            v16h b = frag_b(w2t + lidx * WIDTH, ks, lhalf);
            acc2 = __builtin_amdgcn_wmma_f32_16x16x32_f16(false, a, false, b,
                                                          (short)0, acc2, false, false);
        }
        if (lidx < 6) {
            float bias = b2[lidx];
            #pragma unroll
            for (int rr = 0; rr < 8; ++rr) {
                int grow = mbase + rowbase + rr + 8 * lhalf;
                out6[(size_t)grow * 8 + lidx] = acc2[rr] + bias;
            }
        }
    }
}

// ---------------------------------------------------------------------------
// Volume rendering: one thread per (ray, eps_k); cumulative transmittance
// along samples; atomic mean over K into the output.
// ---------------------------------------------------------------------------
__global__ void zero_out(float* __restrict__ out) {
    int i = blockIdx.x * blockDim.x + threadIdx.x;
    if (i < N_RAYS * 3) out[i] = 0.f;
}

__global__ void render_kernel(const float* __restrict__ den, const float* __restrict__ stdv,
                              const float* __restrict__ out6,
                              const float* __restrict__ eps_den, const float* __restrict__ eps_rgb,
                              float* __restrict__ out) {
    int idx = blockIdx.x * blockDim.x + threadIdx.x;
    if (idx >= N_RAYS * KEPS) return;
    int r = idx / KEPS;
    int k = idx % KEPS;

    float ed  = eps_den[k];
    float er0 = eps_rgb[k * 3 + 0];
    float er1 = eps_rgb[k * 3 + 1];
    float er2 = eps_rgb[k * 3 + 2];

    const float* dr = den  + (size_t)r * N_SAMPLES;
    const float* sr = stdv + (size_t)r * N_SAMPLES;
    const float* o6 = out6 + (size_t)r * N_SAMPLES * 8;

    float T = 1.f, accw = 0.f, c0 = 0.f, c1 = 0.f, c2 = 0.f;
    for (int s = 0; s < N_SAMPLES; ++s) {
        float dk    = dr[s] + sr[s] * ed + ACT_SHIFT;
        float alpha = 1.f - expf(-softplusf(dk) * INTERVAL);
        float w     = alpha * T;
        T *= (1.f - alpha + 1e-10f);

        const float* o = o6 + s * 8;
        float s0 = softplusf(o[3]);
        float s1 = softplusf(o[4]);
        float s2 = softplusf(o[5]);
        c0 += w * sigmoidf(o[0] + s0 * er0);
        c1 += w * sigmoidf(o[1] + s1 * er1);
        c2 += w * sigmoidf(o[2] + s2 * er2);
        accw += w;
    }
    float bg = 1.f - accw;
    const float invK = 1.f / (float)KEPS;
    atomicAdd(&out[r * 3 + 0], (c0 + bg) * invK);
    atomicAdd(&out[r * 3 + 1], (c1 + bg) * invK);
    atomicAdd(&out[r * 3 + 2], (c2 + bg) * invK);
}

// ---------------------------------------------------------------------------
extern "C" void kernel_launch(void* const* d_in, const int* in_sizes, int n_in,
                              void* d_out, int out_size, void* d_ws, size_t ws_size,
                              hipStream_t stream) {
    const float* rays_o  = (const float*)d_in[0];
    const float* rays_d  = (const float*)d_in[1];
    const float* dgrid   = (const float*)d_in[2];
    const float* sgrid   = (const float*)d_in[3];
    const float* k0grid  = (const float*)d_in[4];
    const float* w0      = (const float*)d_in[5];
    const float* b0      = (const float*)d_in[6];
    const float* w1      = (const float*)d_in[7];
    const float* b1      = (const float*)d_in[8];
    const float* w2      = (const float*)d_in[9];
    const float* b2      = (const float*)d_in[10];
    const float* eps_den = (const float*)d_in[11];
    const float* eps_rgb = (const float*)d_in[12];
    float* out = (float*)d_out;

    char* ws = (char*)d_ws;
    size_t off = 0;
    auto carve = [&](size_t bytes) -> void* {
        void* p = ws + off;
        off += (bytes + 255) & ~(size_t)255;
        return p;
    };
    float*    den  = (float*)   carve((size_t)M_TOTAL * sizeof(float));
    float*    stdv = (float*)   carve((size_t)M_TOTAL * sizeof(float));
    float*    out6 = (float*)   carve((size_t)M_TOTAL * 8 * sizeof(float));
    _Float16* w0t  = (_Float16*)carve((size_t)WIDTH * DIM0P * sizeof(_Float16));
    _Float16* w1t  = (_Float16*)carve((size_t)WIDTH * WIDTH * sizeof(_Float16));
    _Float16* w2t  = (_Float16*)carve((size_t)NOUTP * WIDTH * sizeof(_Float16));

    const int prepN = WIDTH * DIM0P + WIDTH * WIDTH + NOUTP * WIDTH;
    prep_weights<<<(prepN + 255) / 256, 256, 0, stream>>>(w0, w1, w2, w0t, w1t, w2t);
    fused_mlp_kernel<<<M_TOTAL / MTILE, 256, 0, stream>>>(rays_o, rays_d, dgrid, sgrid, k0grid,
                                                          w0t, w1t, w2t, b0, b1, b2,
                                                          den, stdv, out6);
    zero_out<<<(N_RAYS * 3 + 255) / 256, 256, 0, stream>>>(out);
    render_kernel<<<(N_RAYS * KEPS + 63) / 64, 64, 0, stream>>>(den, stdv, out6,
                                                                eps_den, eps_rgb, out);
}
